// LSTM_90744069030447
// MI455X (gfx1250) — matrix operand order repaired and checked by reading, not compiled, not run on hardware
//
#include <hip/hip_runtime.h>

typedef __attribute__((ext_vector_type(16))) __bf16 v16bf;
typedef __attribute__((ext_vector_type(8)))  __bf16 v8bf;
typedef __attribute__((ext_vector_type(8)))  float  v8f;

#define HDIM   256
#define FOURH  1024
#define BATCH  64
#define TSTEPS 512
#define ROWS   (BATCH * TSTEPS)   // 32768

// ---------------------------------------------------------------------------
// Pack a [1024 x 256] f32 weight (rows = output col n, cols = k) into the
// CDNA5 16-bit B-matrix (32x16 per tile) register layout, bf16.
// Packed index: ((n_tile*8 + k_tile)*32 + lane)*16 + v
//   lane<16 : N = n_tile*16 + lane,     K = k_tile*32 + (v<8 ? v : 16 + v-8)
//   lane>=16: N = n_tile*16 + lane-16,  K = k_tile*32 + (v<8 ? 8 + v : 24 + v-8)
// ---------------------------------------------------------------------------
__global__ void pack_w(const float* __restrict__ src, __bf16* __restrict__ dst) {
    int idx  = blockIdx.x * blockDim.x + threadIdx.x;   // 0 .. 262143
    int v    = idx & 15;
    int lane = (idx >> 4) & 31;
    int kt   = (idx >> 9) & 7;
    int nt   = idx >> 12;
    int n = nt * 16 + (lane & 15);
    int k = kt * 32 + ((v >> 3) << 4) + ((lane >> 4) << 3) + (v & 7);
    dst[idx] = (__bf16)src[n * HDIM + k];
}

__global__ void f2bf(const float* __restrict__ src, __bf16* __restrict__ dst, int n) {
    int i = blockIdx.x * blockDim.x + threadIdx.x;
    if (i < n) dst[i] = (__bf16)src[i];
}

__global__ void add_bias_vec(const float* __restrict__ a, const float* __restrict__ b,
                             float* __restrict__ out) {
    int i = blockIdx.x * blockDim.x + threadIdx.x;
    if (i < FOURH) out[i] = a[i] + (b ? b[i] : 0.0f);
}

__device__ __forceinline__ v16bf load_a_tile(const __bf16* rowptr, int kb) {
    // A layout for lane: K chunks {kb..kb+7} and {kb+16..kb+23}
    v8bf lo = *(const v8bf*)(rowptr + kb);
    v8bf hi = *(const v8bf*)(rowptr + kb + 16);
    v16bf a;
#pragma unroll
    for (int i = 0; i < 8; ++i) { a[i] = lo[i]; a[8 + i] = hi[i]; }
    return a;
}

// ---------------------------------------------------------------------------
// out = A[ROWS][256](bf16) @ Wp^T + bias, fp32 accumulate.
// Each wave owns a 32M x 64N macro-tile: 2 A tiles x 4 B tiles -> 8 WMMA per
// K-step with only 12 b128 loads (1.5 loads/WMMA), 4x A reuse, 2x B reuse.
// Grid = 2048 blocks x 8 waves = 1024 M-macros x 16 N-macros exactly
// (no bounds branch -> EXEC all ones for WMMA).
// tmajor_out: row r = b*T+t is stored at row t*BATCH+b (xg layout for scan).
// ---------------------------------------------------------------------------
__global__ __launch_bounds__(256) void gemm_bias(
    const __bf16* __restrict__ A,
    const __bf16* __restrict__ Wp,
    const float*  __restrict__ bias,
    float*        __restrict__ out,
    int tmajor_out) {
    int lane   = threadIdx.x & 31;
    int wave   = threadIdx.x >> 5;
    int wid    = blockIdx.x * 8 + wave;
    int tm     = wid >> 4;          // 0..1023  (M macro, 32 rows)
    int tn     = wid & 15;          // 0..15    (N macro, 64 cols = 4 col-tiles)
    int m0     = tm * 32;
    int halfhi = lane >> 4;
    int nl     = lane & 15;
    const __bf16* arow0 = A + (long)(m0 + nl) * HDIM;
    const __bf16* arow1 = arow0 + 16 * HDIM;

    v8f acc[2][4];
#pragma unroll
    for (int m = 0; m < 2; ++m)
#pragma unroll
        for (int j = 0; j < 4; ++j) acc[m][j] = (v8f){};

#pragma unroll
    for (int kt = 0; kt < 8; ++kt) {
        int kb = kt * 32 + (halfhi ? 8 : 0);
        v16bf a0 = load_a_tile(arow0, kb);
        v16bf a1 = load_a_tile(arow1, kb);
#pragma unroll
        for (int j = 0; j < 4; ++j) {
            int ct = tn * 4 + j;    // col tile 0..63
            v16bf b = *(const v16bf*)(Wp + ((long)(ct * 8 + kt) * 32 + lane) * 16);
            acc[0][j] = __builtin_amdgcn_wmma_f32_16x16x32_bf16(false, a0, false, b,
                                                                (short)0, acc[0][j], false, false);
            acc[1][j] = __builtin_amdgcn_wmma_f32_16x16x32_bf16(false, a1, false, b,
                                                                (short)0, acc[1][j], false, false);
        }
    }

    int mb = halfhi << 3;
#pragma unroll
    for (int m = 0; m < 2; ++m) {
#pragma unroll
        for (int j = 0; j < 4; ++j) {
            int   col = (tn * 4 + j) * 16 + nl;
            float bv  = bias[col];
#pragma unroll
            for (int v = 0; v < 8; ++v) {
                int  row  = m0 + m * 16 + mb + v;        // = b*T + t
                long orow = row;
                if (tmajor_out) orow = (long)(row & (TSTEPS - 1)) * BATCH + (row >> 9);
                out[orow * FOURH + col] = acc[m][j][v] + bv;
            }
        }
    }
}

__device__ __forceinline__ float sigm_f(float x) {
    return 1.0f / (1.0f + __expf(-x));
}
__device__ __forceinline__ float tanh_f(float x) {
    return 2.0f / (1.0f + __expf(-2.0f * x)) - 1.0f;
}

// ---------------------------------------------------------------------------
// Sequential LSTM scan. blockIdx.x (0..3) owns batch rows rb..rb+15 for all T.
// Wave u (0..15) owns hidden-unit tile u: gate col-tiles {u, 16+u, 32+u, 48+u}.
// xg is t-major ([T][64][1024]) so each step reads one contiguous 64KB slab.
// xg reads are software-pipelined one step ahead (registers) to keep load
// latency off the serial critical path; each gate uses two independent
// depth-4 WMMA chains merged with VALU adds.
// ---------------------------------------------------------------------------
__global__ __launch_bounds__(512) void lstm_scan(
    const float*  __restrict__ xg,    // [T*64][1024], row = t*64 + b
    const __bf16* __restrict__ Whh,   // packed B layout
    __bf16*       __restrict__ hout)  // [ROWS][256] bf16, row = b*T + t
{
    __shared__ __align__(16) __bf16 hbuf[16 * HDIM];  // 8 KB: h_{t-1}
    int lane   = threadIdx.x & 31;
    int u      = threadIdx.x >> 5;    // unit tile 0..15
    int rb     = blockIdx.x * 16;
    int halfhi = lane >> 4;
    int nl     = lane & 15;
    int mb     = halfhi << 3;
    int col    = u * 16 + nl;         // hidden unit 0..255

    for (int i = threadIdx.x; i < 16 * HDIM; i += blockDim.x) hbuf[i] = (__bf16)0.0f;
    __syncthreads();

    float c_st[8];
#pragma unroll
    for (int v = 0; v < 8; ++v) c_st[v] = 0.0f;

    const __bf16* Bi = Whh + (long)(u)      * 8 * 32 * 16;
    const __bf16* Bf = Whh + (long)(16 + u) * 8 * 32 * 16;
    const __bf16* Bg = Whh + (long)(32 + u) * 8 * 32 * 16;
    const __bf16* Bo = Whh + (long)(48 + u) * 8 * 32 * 16;

    // Prefetch xg for step 0
    v8f nai, naf, nag, nao;
#define LOAD_XG(T_, PI, PF, PG, PO)                                            \
    {                                                                          \
        _Pragma("unroll") for (int v = 0; v < 8; ++v) {                        \
            const float* xr = xg + ((long)(T_) * BATCH + rb + mb + v) * FOURH; \
            (PI)[v] = xr[(u)      * 16 + nl];                                  \
            (PF)[v] = xr[(16 + u) * 16 + nl];                                  \
            (PG)[v] = xr[(32 + u) * 16 + nl];                                  \
            (PO)[v] = xr[(48 + u) * 16 + nl];                                  \
        }                                                                      \
    }
    LOAD_XG(0, nai, naf, nag, nao);

    for (int t = 0; t < TSTEPS; ++t) {
        // Two independent accumulation chains per gate; chain 0 seeded with xg.
        v8f ai0 = nai, af0 = naf, ag0 = nag, ao0 = nao;
        v8f ai1 = (v8f){}, af1 = (v8f){}, ag1 = (v8f){}, ao1 = (v8f){};

        // Prefetch xg for step t+1 (clamped; wait lands in next iteration)
        int tn_ = (t + 1 < TSTEPS) ? (t + 1) : t;
        LOAD_XG(tn_, nai, naf, nag, nao);

#pragma unroll
        for (int kt = 0; kt < 8; ++kt) {
            int kb = kt * 32 + (halfhi ? 8 : 0);
            v16bf a = load_a_tile(hbuf + nl * HDIM, kb);
            long boff = (long)kt * 32 * 16 + lane * 16;
            v16bf bi = *(const v16bf*)(Bi + boff);
            v16bf bf = *(const v16bf*)(Bf + boff);
            v16bf bg = *(const v16bf*)(Bg + boff);
            v16bf bo = *(const v16bf*)(Bo + boff);
            if (kt < 4) {
                ai0 = __builtin_amdgcn_wmma_f32_16x16x32_bf16(false, a, false, bi, (short)0, ai0, false, false);
                af0 = __builtin_amdgcn_wmma_f32_16x16x32_bf16(false, a, false, bf, (short)0, af0, false, false);
                ag0 = __builtin_amdgcn_wmma_f32_16x16x32_bf16(false, a, false, bg, (short)0, ag0, false, false);
                ao0 = __builtin_amdgcn_wmma_f32_16x16x32_bf16(false, a, false, bo, (short)0, ao0, false, false);
            } else {
                ai1 = __builtin_amdgcn_wmma_f32_16x16x32_bf16(false, a, false, bi, (short)0, ai1, false, false);
                af1 = __builtin_amdgcn_wmma_f32_16x16x32_bf16(false, a, false, bf, (short)0, af1, false, false);
                ag1 = __builtin_amdgcn_wmma_f32_16x16x32_bf16(false, a, false, bg, (short)0, ag1, false, false);
                ao1 = __builtin_amdgcn_wmma_f32_16x16x32_bf16(false, a, false, bo, (short)0, ao1, false, false);
            }
        }
        __syncthreads();   // everyone done reading hbuf for step t
#pragma unroll
        for (int v = 0; v < 8; ++v) {
            float ig = sigm_f(ai0[v] + ai1[v]);
            float fg = sigm_f(af0[v] + af1[v]);
            float gg = tanh_f(ag0[v] + ag1[v]);
            float og = sigm_f(ao0[v] + ao1[v]);
            float c  = fg * c_st[v] + ig * gg;
            c_st[v]  = c;
            __bf16 hb = (__bf16)(og * tanh_f(c));
            hbuf[(mb + v) * HDIM + col] = hb;
            hout[((long)(rb + mb + v) * TSTEPS + t) * HDIM + col] = hb;
        }
        __syncthreads();   // h_t visible before step t+1 reads
    }
#undef LOAD_XG
}

// ---------------------------------------------------------------------------
extern "C" void kernel_launch(void* const* d_in, const int* in_sizes, int n_in,
                              void* d_out, int out_size, void* d_ws, size_t ws_size,
                              hipStream_t stream) {
    const float* x    = (const float*)d_in[0];
    const float* wih0 = (const float*)d_in[1];
    const float* whh0 = (const float*)d_in[2];
    const float* bih0 = (const float*)d_in[3];
    const float* bhh0 = (const float*)d_in[4];
    const float* wih1 = (const float*)d_in[5];
    const float* whh1 = (const float*)d_in[6];
    const float* bih1 = (const float*)d_in[7];
    const float* bhh1 = (const float*)d_in[8];
    const float* wfc  = (const float*)d_in[9];
    const float* bfc  = (const float*)d_in[10];

    char*   ws    = (char*)d_ws;
    float*  xg    = (float*)ws;                                        // 128 MiB, reused both layers
    __bf16* bufA  = (__bf16*)(ws + (size_t)ROWS * FOURH * 4);          // 16 MiB: x_bf, later h2_bf
    __bf16* bufB  = bufA + (size_t)ROWS * HDIM;                        // 16 MiB: h1_bf
    __bf16* wpack = bufB + (size_t)ROWS * HDIM;                        // 5 x 512 KiB packed weights
    const size_t WSZ = (size_t)FOURH * HDIM;                           // 262144 elems / weight
    __bf16* wp_ih0 = wpack;
    __bf16* wp_hh0 = wpack + 1 * WSZ;
    __bf16* wp_ih1 = wpack + 2 * WSZ;
    __bf16* wp_hh1 = wpack + 3 * WSZ;
    __bf16* wp_fc  = wpack + 4 * WSZ;
    float*  bias0  = (float*)(wpack + 5 * WSZ);
    float*  bias1  = bias0 + FOURH;
    float*  biasf  = bias0 + 2 * FOURH;

    // Pack weights into WMMA B-operand layout (bf16)
    pack_w<<<1024, 256, 0, stream>>>(wih0, wp_ih0);
    pack_w<<<1024, 256, 0, stream>>>(whh0, wp_hh0);
    pack_w<<<1024, 256, 0, stream>>>(wih1, wp_ih1);
    pack_w<<<1024, 256, 0, stream>>>(whh1, wp_hh1);
    pack_w<<<1024, 256, 0, stream>>>(wfc,  wp_fc);

    add_bias_vec<<<4, 256, 0, stream>>>(bih0, bhh0, bias0);
    add_bias_vec<<<4, 256, 0, stream>>>(bih1, bhh1, bias1);
    add_bias_vec<<<4, 256, 0, stream>>>(bfc,  nullptr, biasf);

    // Layer 0
    f2bf<<<(ROWS * HDIM + 255) / 256, 256, 0, stream>>>(x, bufA, ROWS * HDIM);
    gemm_bias<<<2048, 256, 0, stream>>>(bufA, wp_ih0, bias0, xg, 1);   // xg0 (t-major)
    lstm_scan<<<4, 512, 0, stream>>>(xg, wp_hh0, bufB);                // h1 (b-major)

    // Layer 1
    gemm_bias<<<2048, 256, 0, stream>>>(bufB, wp_ih1, bias1, xg, 1);   // xg1 (t-major)
    lstm_scan<<<4, 512, 0, stream>>>(xg, wp_hh1, bufA);                // h2 (b-major)

    // Final FC (b-major output as the reference requires)
    gemm_bias<<<2048, 256, 0, stream>>>(bufA, wp_fc, biasf, (float*)d_out, 0);
}
